// Attention_49624052138026
// MI455X (gfx1250) — matrix operand order repaired and checked by reading, not compiled, not run on hardware
//
#include <hip/hip_runtime.h>

// ---------------------------------------------------------------------------
// Single-head attention fwd for MI455X (gfx1250, wave32, WMMA).
//   q = x@Wq; k = x@Wk; v = x@Wv
//   wei = softmax(q k^T / sqrt(1024)); dropout(0.2); out = wei @ v
// Strategy: bf16 WMMA (v_wmma_f32_16x16x32_bf16) everywhere, f32 accumulate.
// Kernel 1 stages tiles with GLOBAL_LOAD_ASYNC_TO_LDS (double-buffered,
// S_WAIT_ASYNCCNT); f32->bf16 via compiler fptrunc (native cvt if available).
// Workspace layout (d_ws, needs ~96.2 MB):
//   [0)        Q  bf16 [8*2048][1024] row-major          (32 MB)
//   [32MB)     K  bf16 [8*2048][1024] row-major          (32 MB)
//   [64MB)     Vt bf16 [8][1024][2048]  (V transposed)   (32 MB)
//   [96MB)     stats f32 [8*2048][2]  (row max m, row sumexp l)
// ---------------------------------------------------------------------------

typedef __attribute__((ext_vector_type(16))) __bf16 bf16x16;
typedef __attribute__((ext_vector_type(2)))  __bf16 bf16x2;
typedef __attribute__((ext_vector_type(8)))  float  f32x8;
typedef __attribute__((ext_vector_type(4)))  int    i32x4;

union Frag { bf16x16 v; uint4 q[2]; };

#define GAS __attribute__((address_space(1)))   // global
#define LAS __attribute__((address_space(3)))   // LDS

#if __has_builtin(__builtin_amdgcn_global_load_async_to_lds_b128) && \
    __has_builtin(__builtin_amdgcn_global_load_async_to_lds_b32)  && \
    __has_builtin(__builtin_amdgcn_s_wait_asynccnt)
#define USE_ASYNC_LDS 1
#else
#define USE_ASYNC_LDS 0
#endif

#if __has_builtin(__builtin_amdgcn_cvt_pk_bf16_f32)
#define HAS_PK_BF16 1
#else
#define HAS_PK_BF16 0
#endif

// Single f32 -> bf16 bits via compiler fptrunc (native v_cvt if the target
// has one; otherwise LLVM emits the RNE fixup sequence).
__device__ __forceinline__ unsigned short bf_bits(float f) {
  union { __bf16 h; unsigned short u; } c;
  c.h = (__bf16)f;
  return c.u;
}

// Two f32 -> packed bf16 pair in one 32-bit lane.
__device__ __forceinline__ unsigned pack_bf16(float a, float b) {
#if HAS_PK_BF16
  union { bf16x2 v; unsigned u; } c;
  c.v = __builtin_amdgcn_cvt_pk_bf16_f32(a, b);
  return c.u;
#else
  union { bf16x2 v; unsigned u; } c;
  bf16x2 t = { (__bf16)a, (__bf16)b };
  c.v = t;
  return c.u;
#endif
}

__device__ __forceinline__ f32x8 zero8() {
  f32x8 z = {0.f,0.f,0.f,0.f,0.f,0.f,0.f,0.f};
  return z;
}

#define WMMA_BF16(A,B,C) \
  __builtin_amdgcn_wmma_f32_16x16x32_bf16(false,(A),false,(B),(short)0,(C),false,false)

// Deterministic dropout mask (structural stand-in for jax bernoulli).
__device__ __forceinline__ bool keep_mask(int b, int t, int s) {
  unsigned x = (unsigned)(t * 2048 + s) ^ ((unsigned)b * 0x9E3779B9u);
  x ^= x >> 16; x *= 0x85EBCA6Bu; x ^= x >> 13; x *= 0xC2B2AE35u; x ^= x >> 16;
  return x < 3435973836u;        // keep_p = 0.8
}

// Load 2x8 contiguous f32 runs from LDS, convert into a bf16 fragment.
__device__ __forceinline__ bf16x16 ld_cvt16(const float* base, int o0, int o1) {
  union { bf16x16 v; __bf16 e[16]; } f;
#pragma unroll
  for (int j = 0; j < 8; ++j) f.e[j]     = (__bf16)base[o0 + j];
#pragma unroll
  for (int j = 0; j < 8; ++j) f.e[8 + j] = (__bf16)base[o1 + j];
  return f.v;
}

// ---------------------------------------------------------------------------
// Kernel 1: QKV projection GEMM  [16384 x 1024] = x[16384 x 1024] * W[1024 x 1024]
// grid (128, 8, 3): z selects Wq/Wk/Wv. 256 thr = 8 waves; wave tile 32x64.
// Tiles staged raw-f32 into LDS: x via async b128 (natural layout), W via
// async b32 *scattered transposed* (per-lane LDS dest makes this free).
// 20 async ops/thread/tile -> s_wait_asynccnt(20) waits exactly one tile.
// V output is stored transposed (Vt[b][e][t]) so the PV B-operand is contiguous.
// ---------------------------------------------------------------------------
#define K1_ASYNC_OPS 20

__device__ __forceinline__ void stage_tile(float (*xs)[36], float (*wt)[36],
                                           const float* __restrict__ x,
                                           const float* __restrict__ W,
                                           int gm0, int gn0, int k0, int tid)
{
#if USE_ASYNC_LDS
#pragma unroll
  for (int p = 0; p < 4; ++p) {                 // x tile 128x32 f32, 16B/lane
    const int r = (tid >> 3) + p * 32;
    const int c = (tid & 7) * 4;
    __builtin_amdgcn_global_load_async_to_lds_b128(
        (GAS i32x4*)(void*)(x + (size_t)(gm0 + r) * 1024 + k0 + c),
        (LAS i32x4*)(void*)&xs[r][c], 0, 0);
  }
#pragma unroll
  for (int p = 0; p < 16; ++p) {                // W tile 32x128 f32, transposed
    const int rr = (tid >> 7) * 16 + p;
    const int cc = tid & 127;
    __builtin_amdgcn_global_load_async_to_lds_b32(
        (GAS int*)(void*)(W + (size_t)(k0 + rr) * 1024 + gn0 + cc),
        (LAS int*)(void*)&wt[cc][rr], 0, 0);
  }
#else
#pragma unroll
  for (int p = 0; p < 4; ++p) {
    const int r = (tid >> 3) + p * 32;
    const int c = (tid & 7) * 4;
    *(float4*)&xs[r][c] = *(const float4*)(x + (size_t)(gm0 + r) * 1024 + k0 + c);
  }
#pragma unroll
  for (int p = 0; p < 16; ++p) {
    const int rr = (tid >> 7) * 16 + p;
    const int cc = tid & 127;
    wt[cc][rr] = W[(size_t)(k0 + rr) * 1024 + gn0 + cc];
  }
#endif
}

__global__ __launch_bounds__(256)
void qkv_gemm(const float* __restrict__ x,
              const float* __restrict__ Wq,
              const float* __restrict__ Wk,
              const float* __restrict__ Wv,
              unsigned short* __restrict__ qb,
              unsigned short* __restrict__ kb,
              unsigned short* __restrict__ vt)
{
  __shared__ float xsf[2][128][36];   // raw f32 A tiles (stride 144B, 16B aligned)
  __shared__ float wtTf[2][128][36];  // raw f32 B tiles, transposed [n][k]

  const int tid  = threadIdx.x;
  const int lane = tid & 31, wave = tid >> 5;
  const int h = lane >> 4, li = lane & 15;
  const int z = blockIdx.z;
  const float* W = (z == 0) ? Wq : ((z == 1) ? Wk : Wv);
  const int gm0 = blockIdx.x * 128;
  const int gn0 = blockIdx.y * 128;
  const int mbase = (wave & 3) * 32;        // 4 waves along M
  const int nbase = (wave >> 2) * 64;       // 2 waves along N

  f32x8 acc[2][4];
#pragma unroll
  for (int mt = 0; mt < 2; ++mt)
#pragma unroll
    for (int nt = 0; nt < 4; ++nt) acc[mt][nt] = zero8();

  stage_tile(xsf[0], wtTf[0], x, W, gm0, gn0, 0, tid);

  for (int kt = 0; kt < 32; ++kt) {
    const int cur = kt & 1;
    if (kt + 1 < 32)
      stage_tile(xsf[cur ^ 1], wtTf[cur ^ 1], x, W, gm0, gn0, (kt + 1) * 32, tid);
#if USE_ASYNC_LDS
    if (kt + 1 < 32) __builtin_amdgcn_s_wait_asynccnt(K1_ASYNC_OPS);
    else             __builtin_amdgcn_s_wait_asynccnt(0);
#endif
    __syncthreads();   // current tile visible to all waves

    // Fragments in contiguous k order (k = 16*h + s) for A and B, cvt f32->bf16 inline.
    bf16x16 a[2], bfr[4];
#pragma unroll
    for (int mt = 0; mt < 2; ++mt)
      a[mt] = ld_cvt16(&xsf[cur][mbase + mt * 16 + li][0], h * 16, h * 16 + 8);
#pragma unroll
    for (int nt = 0; nt < 4; ++nt)
      bfr[nt] = ld_cvt16(&wtTf[cur][nbase + nt * 16 + li][0], h * 16, h * 16 + 8);
#pragma unroll
    for (int mt = 0; mt < 2; ++mt)
#pragma unroll
      for (int nt = 0; nt < 4; ++nt)
        acc[mt][nt] = WMMA_BF16(a[mt], bfr[nt], acc[mt][nt]);
    __syncthreads();   // all reads of `cur` done before it is re-staged
  }

  // Store bf16. C/D layout: VGPR i -> row i + 8*h, lane -> col.
  // z is block-uniform: one branch, straight-line store loops.
  if (z != 2) {
    unsigned short* __restrict__ dst = (z == 0) ? qb : kb;
#pragma unroll
    for (int mt = 0; mt < 2; ++mt)
#pragma unroll
      for (int nt = 0; nt < 4; ++nt)
#pragma unroll
        for (int i = 0; i < 8; ++i) {
          const int gr = gm0 + mbase + mt * 16 + i + h * 8;
          const int gc = gn0 + nbase + nt * 16 + li;
          dst[(size_t)gr * 1024 + gc] = bf_bits(acc[mt][nt][i]);
        }
  } else {
#pragma unroll
    for (int mt = 0; mt < 2; ++mt)
#pragma unroll
      for (int nt = 0; nt < 4; ++nt)
#pragma unroll
        for (int i = 0; i < 8; ++i) {
          const int gr = gm0 + mbase + mt * 16 + i + h * 8;
          const int gc = gn0 + nbase + nt * 16 + li;
          const int bb = gr >> 11, t = gr & 2047;
          vt[((size_t)bb * 1024 + gc) * 2048 + t] = bf_bits(acc[mt][nt][i]);
        }
  }
}

// ---------------------------------------------------------------------------
// Kernel 2: softmax stats pass. Workgroup = (batch, 16 query rows).
// 8 waves each own 256 keys; online (m, l) per row; combine in LDS.
// ---------------------------------------------------------------------------
__global__ __launch_bounds__(256)
void attn_stats(const unsigned short* __restrict__ qb,
                const unsigned short* __restrict__ kb,
                float* __restrict__ stats)
{
  const int tid = threadIdx.x, lane = tid & 31, wave = tid >> 5;
  const int b  = blockIdx.x >> 7;
  const int q0 = (blockIdx.x & 127) * 16;
  const int h = lane >> 4, li = lane & 15;
  const float scale = 0.03125f;  // 1024^-0.5
  const size_t qrow = ((size_t)b * 2048 + q0 + li) * 1024;

  float m[8], l[8];
#pragma unroll
  for (int i = 0; i < 8; ++i) { m[i] = -3.0e38f; l[i] = 0.f; }
  __shared__ float sm[8][16], sl[8][16];

  for (int s0 = wave * 256; s0 < wave * 256 + 256; s0 += 32) {
    f32x8 acc0 = zero8(), acc1 = zero8();
    const size_t krow0 = ((size_t)b * 2048 + s0 + li) * 1024;
    const size_t krow1 = krow0 + (size_t)16 * 1024;
#pragma unroll 4
    for (int kd = 0; kd < 1024; kd += 32) {
      Frag a, b0, b1;
      a.q[0]  = *(const uint4*)(qb + qrow  + kd + h * 16);
      a.q[1]  = *(const uint4*)(qb + qrow  + kd + h * 16 + 8);
      b0.q[0] = *(const uint4*)(kb + krow0 + kd + h * 16);
      b0.q[1] = *(const uint4*)(kb + krow0 + kd + h * 16 + 8);
      b1.q[0] = *(const uint4*)(kb + krow1 + kd + h * 16);
      b1.q[1] = *(const uint4*)(kb + krow1 + kd + h * 16 + 8);
      acc0 = WMMA_BF16(a.v, b0.v, acc0);
      acc1 = WMMA_BF16(a.v, b1.v, acc1);
    }
    // Online softmax update; rows live across VGPR index, cols across lanes.
#pragma unroll
    for (int i = 0; i < 8; ++i) {
      const float v0 = acc0[i] * scale, v1 = acc1[i] * scale;
      float mx = fmaxf(v0, v1);
      for (int d = 1; d < 16; d <<= 1) mx = fmaxf(mx, __shfl_xor(mx, d, 32));
      const float mn = fmaxf(m[i], mx);
      float e = __expf(v0 - mn) + __expf(v1 - mn);
      for (int d = 1; d < 16; d <<= 1) e += __shfl_xor(e, d, 32);
      l[i] = l[i] * __expf(m[i] - mn) + e;
      m[i] = mn;
    }
  }
  if ((lane & 15) == 0) {
#pragma unroll
    for (int i = 0; i < 8; ++i) { sm[wave][h * 8 + i] = m[i]; sl[wave][h * 8 + i] = l[i]; }
  }
  __syncthreads();
  if (tid < 16) {
    float M = -3.0e38f, L = 0.f;
    for (int w = 0; w < 8; ++w) M = fmaxf(M, sm[w][tid]);
    for (int w = 0; w < 8; ++w) L += sl[w][tid] * __expf(sm[w][tid] - M);
    const size_t r = (size_t)b * 2048 + q0 + tid;
    stats[r * 2] = M; stats[r * 2 + 1] = L;
  }
}

// ---------------------------------------------------------------------------
// Kernel 3: P = dropout(exp(S - m)/l) ; out = P @ V.
// Workgroup = (batch, 16 query rows). Per 256-key tile: each wave builds its
// 16x32 slab of P (bf16, to LDS), then accumulates its private 128-wide E
// slice (8 C-tiles = 64 accum VGPRs) against Vt.
// ---------------------------------------------------------------------------
__global__ __launch_bounds__(256)
void attn_pv(const unsigned short* __restrict__ qb,
             const unsigned short* __restrict__ kb,
             const unsigned short* __restrict__ vt,
             const float* __restrict__ stats,
             float* __restrict__ out)
{
  const int tid = threadIdx.x, lane = tid & 31, wave = tid >> 5;
  const int b  = blockIdx.x >> 7;
  const int q0 = (blockIdx.x & 127) * 16;
  const int h = lane >> 4, li = lane & 15;
  const int e0 = wave * 128;
  const float scale = 0.03125f;
  const float inv_keep = 1.25f;             // 1 / 0.8

  __shared__ unsigned short P[16][264];     // 16 x 256 bf16, padded (528B stride)

  f32x8 o[8];
#pragma unroll
  for (int n = 0; n < 8; ++n) o[n] = zero8();

  float Mr[8], iLr[8];
#pragma unroll
  for (int i = 0; i < 8; ++i) {
    const size_t r = (size_t)b * 2048 + q0 + i + h * 8;
    Mr[i]  = stats[r * 2];
    iLr[i] = 1.0f / stats[r * 2 + 1];
  }
  const size_t qrow = ((size_t)b * 2048 + q0 + li) * 1024;

  for (int kt = 0; kt < 8; ++kt) {
    const int sbase = kt * 256;
    const int sw = sbase + wave * 32;

    // Warm L2/L0 for the next key tile's Vt rows (global_prefetch_b8).
    if (kt + 1 < 8) {
      const unsigned short* pf =
          vt + ((size_t)b * 1024 + e0 + (lane & 63)) * 2048 + (size_t)(kt + 1) * 256;
      __builtin_prefetch(pf, 0, 1);
    }

    f32x8 acc0 = zero8(), acc1 = zero8();
    const size_t krow0 = ((size_t)b * 2048 + sw + li) * 1024;
    const size_t krow1 = krow0 + (size_t)16 * 1024;
#pragma unroll 4
    for (int kd = 0; kd < 1024; kd += 32) {
      Frag a, b0, b1;
      a.q[0]  = *(const uint4*)(qb + qrow  + kd + h * 16);
      a.q[1]  = *(const uint4*)(qb + qrow  + kd + h * 16 + 8);
      b0.q[0] = *(const uint4*)(kb + krow0 + kd + h * 16);
      b0.q[1] = *(const uint4*)(kb + krow0 + kd + h * 16 + 8);
      b1.q[0] = *(const uint4*)(kb + krow1 + kd + h * 16);
      b1.q[1] = *(const uint4*)(kb + krow1 + kd + h * 16 + 8);
      acc0 = WMMA_BF16(a.v, b0.v, acc0);
      acc1 = WMMA_BF16(a.v, b1.v, acc1);
    }
    // exp + normalize + dropout -> bf16 P in LDS.
#pragma unroll
    for (int i = 0; i < 8; ++i) {
      const int r = i + h * 8;
      const int t = q0 + r;
      const int s0 = sw + li;
      const int s1 = sw + 16 + li;
      float p0 = __expf(acc0[i] * scale - Mr[i]) * iLr[i];
      float p1 = __expf(acc1[i] * scale - Mr[i]) * iLr[i];
      p0 = keep_mask(b, t, s0) ? p0 * inv_keep : 0.f;
      p1 = keep_mask(b, t, s1) ? p1 * inv_keep : 0.f;
      P[r][wave * 32 + li]      = bf_bits(p0);
      P[r][wave * 32 + 16 + li] = bf_bits(p1);
    }
    __syncthreads();
    // o += P(16x256) @ Vt-slice(256x128)
#pragma unroll
    for (int kc = 0; kc < 8; ++kc) {
      Frag a;
      const unsigned short* prow = &P[li][kc * 32];
      a.q[0] = *(const uint4*)(prow + h * 16);
      a.q[1] = *(const uint4*)(prow + h * 16 + 8);
      const size_t scol = (size_t)sbase + kc * 32 + h * 16;
#pragma unroll
      for (int n = 0; n < 8; ++n) {
        Frag bv;
        const unsigned short* vp =
            vt + ((size_t)b * 1024 + e0 + n * 16 + li) * 2048 + scol;
        bv.q[0] = *(const uint4*)(vp);
        bv.q[1] = *(const uint4*)(vp + 8);
        o[n] = WMMA_BF16(a.v, bv.v, o[n]);
      }
    }
    __syncthreads();
  }

  // f32 output store.
#pragma unroll
  for (int n = 0; n < 8; ++n)
#pragma unroll
    for (int i = 0; i < 8; ++i) {
      const int t = q0 + i + h * 8;
      const int e = e0 + n * 16 + li;
      out[((size_t)b * 2048 + t) * 1024 + e] = o[n][i];
    }
}

// ---------------------------------------------------------------------------
extern "C" void kernel_launch(void* const* d_in, const int* in_sizes, int n_in,
                              void* d_out, int out_size, void* d_ws, size_t ws_size,
                              hipStream_t stream) {
  const float* x  = (const float*)d_in[0];
  const float* Wq = (const float*)d_in[1];
  const float* Wk = (const float*)d_in[2];
  const float* Wv = (const float*)d_in[3];

  const size_t NTOK = (size_t)8 * 2048;          // 16384 rows
  unsigned short* qb = (unsigned short*)d_ws;
  unsigned short* kb = qb + NTOK * 1024;
  unsigned short* vt = kb + NTOK * 1024;
  float* stats = (float*)(vt + NTOK * 1024);
  float* out = (float*)d_out;

  dim3 g1(128, 8, 3);                             // M/128, N/128, {Q,K,V}
  qkv_gemm<<<g1, 256, 0, stream>>>(x, Wq, Wk, Wv, qb, kb, vt);
  attn_stats<<<1024, 256, 0, stream>>>(qb, kb, stats);
  attn_pv<<<1024, 256, 0, stream>>>(qb, kb, vt, stats, out);
}
